// DualTokenQFormer_23098334118044
// MI455X (gfx1250) — matrix operand order repaired and checked by reading, not compiled, not run on hardware
//
#include <hip/hip_runtime.h>

// ---------------------------------------------------------------------------
// CDNA5 (gfx1250) wave32 + WMMA bf16 implementation of DualTokenQFormer.
// Single-query cross-attention linearized:  scores via QK = wk_h^T qp_h,
// output via weighted token-sum C then wv apply.  Streaming passes are
// flash-style with online softmax; chunks are staged into LDS with
// GLOBAL_LOAD_ASYNC_TO_LDS_B128 (ASYNCcnt), operands converted with native
// v_cvt_pk_bf16_f32, all matrix math in v_wmma_f32_16x16x32_bf16.
// ---------------------------------------------------------------------------

typedef __attribute__((ext_vector_type(16))) __bf16 v16bf;
typedef __attribute__((ext_vector_type(8)))  float  v8f;

#define NSEG 16   // segments per batch for the streaming cross-attn

static __device__ __forceinline__ __bf16 f2bf(float f) {
  return (__bf16)f;                       // native v_cvt (RNE) on gfx1250
}
static __device__ __forceinline__ unsigned short f2bf_bits(float f) {
  __bf16 h = (__bf16)f;
  return __builtin_bit_cast(unsigned short, h);
}

static __device__ __forceinline__ v8f wmma_bf16(v16bf a, v16bf b, v8f c) {
  return __builtin_amdgcn_wmma_f32_16x16x32_bf16(false, a, false, b, (short)0, c,
                                                 false, false);
}

// 16 contiguous bf16 (LDS or global generic pointers; 16B aligned).
// Packed uint4 pairs are already in v16bf register layout -> pure bit_cast.
struct U32x8 { uint4 a, b; };
static __device__ __forceinline__ v16bf ld16bf(const unsigned short* p) {
  U32x8 u;
  u.a = *(const uint4*)p;
  u.b = *(const uint4*)(p + 8);
  return __builtin_bit_cast(v16bf, u);
}
static __device__ __forceinline__ void cvt8(v16bf& v, int base, float4 a, float4 b) {
  v[base + 0] = f2bf(a.x); v[base + 1] = f2bf(a.y);
  v[base + 2] = f2bf(a.z); v[base + 3] = f2bf(a.w);
  v[base + 4] = f2bf(b.x); v[base + 5] = f2bf(b.y);
  v[base + 6] = f2bf(b.z); v[base + 7] = f2bf(b.w);
}

// low 32 bits of a generic pointer to a __shared__ object == LDS byte offset
static __device__ __forceinline__ unsigned lds_off(const void* p) {
  return (unsigned)(unsigned long long)p;
}

// ---------------------------------------------------------------------------
// Generic skinny WMMA GEMM:  C[M,N] = alpha * (A[M,K] x B) + bias, optional GELU
//   transW=1 : B = W^T with W row-major (N x K)      (torch Linear weights)
//   transW=0 : B = W    with W row-major (K x N)
// One wave per 16x16 C tile. grid = (M/16, N/16, Z) with per-Z strides.
// ---------------------------------------------------------------------------
__global__ __launch_bounds__(32) void gemm_wmma(
    const float* __restrict__ A, int lda, long aZ,
    const float* __restrict__ W, int ldw, long wZ, int transW,
    const float* __restrict__ bias, int bZ,
    float* __restrict__ Cf, unsigned short* __restrict__ Cb, int ldc, long cZ,
    int K, float alpha, int gelu)
{
  const int mt = blockIdx.x, nt = blockIdx.y, z = blockIdx.z;
  const int lane = threadIdx.x;
  const int half = lane >> 4, l16 = lane & 15;
  const float* Ab = A + (long)z * aZ;
  const float* Wb = W + (long)z * wZ;
  const int arow = mt * 16 + l16;   // A operand: lanes 0-15 and 16-31 both carry rows 0..15
  const int ncol = nt * 16 + l16;   // B operand: lane -> column

  v8f acc;
  #pragma unroll
  for (int r = 0; r < 8; ++r) acc[r] = 0.0f;

  for (int k0 = 0; k0 < K; k0 += 32) {
    v16bf a;
    {
      // lane<16: K 0..15; lane>=16: K 16..31
      const float* ap = Ab + (long)arow * lda + k0 + half * 16;
      cvt8(a, 0, *(const float4*)(ap +  0), *(const float4*)(ap +  4));
      cvt8(a, 8, *(const float4*)(ap +  8), *(const float4*)(ap + 12));
    }
    v16bf bb;
    if (transW) {
      // lane<16: K 0..15 ; lane>=16: K 16..31 (down column ncol of W^T == row ncol of W)
      const float* wp = Wb + (long)ncol * ldw + k0 + half * 16;
      cvt8(bb, 0, *(const float4*)(wp + 0), *(const float4*)(wp + 4));
      cvt8(bb, 8, *(const float4*)(wp + 8), *(const float4*)(wp + 12));
    } else {
      const int kb = k0 + half * 16;
      #pragma unroll
      for (int j = 0; j < 16; ++j)
        bb[j] = f2bf(Wb[(long)(kb + j) * ldw + ncol]);
    }
    acc = wmma_bf16(a, bb, acc);
  }

  const float bv = bias ? bias[(long)z * bZ + ncol] : 0.0f;
  #pragma unroll
  for (int r = 0; r < 8; ++r) {
    const int row = mt * 16 + r + 8 * half;
    float v = acc[r] * alpha + bv;
    if (gelu) v = 0.5f * v * (1.0f + erff(v * 0.70710678118654752f));
    const long off = (long)z * cZ + (long)row * ldc + ncol;
    if (Cb) Cb[off] = f2bf_bits(v);
    else    Cf[off] = v;
  }
}

// ---------------------------------------------------------------------------
// Fused single-query cross-attention streaming pass (flash-style, WMMA bf16).
// grid = (B, NSEG), block = 256 (8 waves).  Per 32-token chunk:
//   async-DMA the raw fp32 chunk into LDS (row-major, padded stride 1028);
//   S^T = X @ QK^T (waves split token-tile x dim-quarter);
//   online softmax per head;  C(head,dim) += W @ X  (waves split dims).
// Emits per-(b,seg) partials (running max m, running sum s, C 16x1024).
// ---------------------------------------------------------------------------
#define XSTRIDE 1028   // 1024 + 4 float pad: b128 row reads spread across banks

__global__ __launch_bounds__(256) void cross_stream(
    const float* __restrict__ X, int Ntok,
    const unsigned short* __restrict__ QK,      // [B][16][1024] bf16 (scale folded)
    float* __restrict__ pc, float* __restrict__ pm, float* __restrict__ ps)
{
  __shared__ __align__(16) float Xf[32 * XSTRIDE];        // fp32 chunk, ~131.5 KB
  __shared__ float Sp[8 * 256];                           // per-wave partial S^T tiles
  __shared__ float Sfin[32 * 16];                         // S[token][head]
  __shared__ __align__(16) unsigned short Ws[16 * 32];    // exp-weights [head][tok] bf16
  __shared__ float scaleL[16];

  const int b = blockIdx.x, seg = blockIdx.y;
  const int tid = threadIdx.x;
  const int lane = tid & 31, wave = tid >> 5;
  const int half = lane >> 4, l16 = lane & 15;
  const int seglen = Ntok / NSEG;
  const int tok0 = seg * seglen;
  const float* Xb = X + (long)b * Ntok * 1024;
  const unsigned short* QKb = QK + (long)b * 16 * 1024;

  v8f acc[8];
  #pragma unroll
  for (int t = 0; t < 8; ++t)
    #pragma unroll
    for (int r = 0; r < 8; ++r) acc[t][r] = 0.0f;

  float Mrun = -1e30f, Srun = 0.0f;   // valid on wave 0 lanes 0..15 (per head)

  for (int c0 = tok0; c0 < tok0 + seglen; c0 += 32) {
    // ---- async DMA: 32 tokens x 1024 fp32 -> LDS (ASYNCcnt-tracked, no VGPR data)
    for (int i = tid; i < 32 * 256; i += 256) {        // 8192 x float4
      const int t = i >> 8;
      const int d4 = (i & 255) << 2;
      unsigned long long ga =
          (unsigned long long)(Xb + (long)(c0 + t) * 1024 + d4);
      unsigned la = lds_off(&Xf[t * XSTRIDE + d4]);
      asm volatile("global_load_async_to_lds_b128 %0, %1, off"
                   :: "v"(la), "v"(ga) : "memory");
    }
    asm volatile("s_wait_asynccnt 0x0" ::: "memory");
    __syncthreads();

    // ---- partial scores: wave -> (token-tile = wave&1, dim-quarter = wave>>1)
    {
      const int tile = wave & 1;
      const int dq = wave >> 1;
      const int tok = tile * 16 + l16;
      const float* xrow = &Xf[tok * XSTRIDE];
      v8f sc;
      #pragma unroll
      for (int r = 0; r < 8; ++r) sc[r] = 0.0f;
      for (int kk = 0; kk < 8; ++kk) {
        const int kbase = dq * 256 + kk * 32;
        const int dofs = half * 16;
        const float* ap = xrow + kbase + dofs;
        v16bf a;   // A = X tile (16 tok x 32 dims): contiguous LDS b128 + cvt
        cvt8(a, 0, *(const float4*)(ap +  0), *(const float4*)(ap +  4));
        cvt8(a, 8, *(const float4*)(ap +  8), *(const float4*)(ap + 12));
        // B = QK^T (32 dims x 16 heads): lane = head, contiguous bf16 (bit_cast)
        v16bf bq = ld16bf(QKb + (long)l16 * 1024 + kbase + half * 16);
        sc = wmma_bf16(a, bq, sc);
      }
      #pragma unroll
      for (int r = 0; r < 8; ++r) Sp[wave * 256 + r * 32 + lane] = sc[r];
    }
    __syncthreads();

    // ---- reduce the 4 dim-quarter partials -> Sfin[token][head]
    for (int i = tid; i < 512; i += 256) {
      const int token = i >> 4, head = i & 15;
      const int tile = token >> 4, tl = token & 15;
      const int r = tl & 7, hh = tl >> 3;
      const int lidx = head + (hh << 4);
      float s = 0.0f;
      #pragma unroll
      for (int dq = 0; dq < 4; ++dq) s += Sp[(dq * 2 + tile) * 256 + r * 32 + lidx];
      Sfin[token * 16 + head] = s;
    }
    __syncthreads();

    // ---- online softmax bookkeeping (wave 0, one lane per head)
    if (wave == 0 && lane < 16) {
      float mc = -1e30f;
      for (int t = 0; t < 32; ++t) mc = fmaxf(mc, Sfin[t * 16 + lane]);
      const float Mn = fmaxf(Mrun, mc);
      const float scl = __expf(Mrun - Mn);
      float sum = Srun * scl;
      for (int t = 0; t < 32; ++t) {
        const float wgt = __expf(Sfin[t * 16 + lane] - Mn);
        sum += wgt;
        Ws[lane * 32 + t] = f2bf_bits(wgt);
      }
      Srun = sum; Mrun = Mn; scaleL[lane] = scl;
    }
    __syncthreads();

    // ---- C(head, dim) = C*scale + W(16x32) @ X(32 x 128-dim slice per wave)
    {
      float scl[8];
      #pragma unroll
      for (int r = 0; r < 8; ++r) scl[r] = scaleL[r + (half << 3)];
      const int dofs = half * 16;
      // A = exp-weights, lane = head, contiguous bf16 rows (bit_cast)
      U32x8 uw;
      uw.a = *(const uint4*)(Ws + l16 * 32 + dofs);
      uw.b = *(const uint4*)(Ws + l16 * 32 + 8 + dofs);
      const v16bf aw = __builtin_bit_cast(v16bf, uw);
      #pragma unroll
      for (int t = 0; t < 8; ++t) {
        #pragma unroll
        for (int r = 0; r < 8; ++r) acc[t][r] *= scl[r];
        const int dim = wave * 128 + t * 16 + l16;
        v16bf xb;   // B: lane = dim, K = tokens (strided fp32 gather + cvt)
        #pragma unroll
        for (int j = 0; j < 16; ++j)
          xb[j] = f2bf(Xf[(half * 16 + j) * XSTRIDE + dim]);
        acc[t] = wmma_bf16(aw, xb, acc[t]);
      }
    }
    __syncthreads();
  }

  // ---- write segment partials
  const long base = (long)(b * NSEG + seg) * 16;
  #pragma unroll
  for (int t = 0; t < 8; ++t)
    #pragma unroll
    for (int r = 0; r < 8; ++r) {
      const int head = r + 8 * half;
      const int dim = wave * 128 + t * 16 + l16;
      pc[(base + head) * 1024 + dim] = acc[t][r];
    }
  if (wave == 0 && lane < 16) { pm[base + lane] = Mrun; ps[base + lane] = Srun; }
}

// ---------------------------------------------------------------------------
// small helper kernels
// ---------------------------------------------------------------------------
__global__ void mean_partial(const float* __restrict__ X, float* __restrict__ pmean) {
  const int b = blockIdx.x, c = blockIdx.y;
  const int d = blockIdx.z * 256 + threadIdx.x;
  const float* p = X + ((long)b * 4096 + (long)c * 512) * 1024 + d;
  float s = 0.0f;
  for (int t = 0; t < 512; ++t) s += p[(long)t * 1024];
  pmean[((long)b * 8 + c) * 1024 + d] = s;
}

__global__ void q_init(const float* __restrict__ pmean, const float* __restrict__ maxini,
                       const float* __restrict__ qbase, const float* __restrict__ rolew,
                       const float* __restrict__ timew, const int* __restrict__ fidx,
                       float* __restrict__ q) {
  const int b = blockIdx.x;
  int fi = fidx[b]; fi = fi < 0 ? 0 : (fi > 4095 ? 4095 : fi);
  for (int d = threadIdx.x; d < 1024; d += blockDim.x) {
    float m = 0.0f;
    for (int c = 0; c < 8; ++c) m += pmean[((long)b * 8 + c) * 1024 + d];
    m *= (1.0f / 4096.0f);
    const float te = timew[(long)fi * 1024 + d];
    q[((long)b * 2 + 0) * 1024 + d] = m + qbase[d] + rolew[d] + te;
    q[((long)b * 2 + 1) * 1024 + d] =
        maxini[(long)b * 1024 + d] + qbase[1024 + d] + rolew[1024 + d] + te;
  }
}

__global__ void ln_rows(const float* __restrict__ x, const float* __restrict__ g,
                        const float* __restrict__ b, float* __restrict__ y) {
  const int row = blockIdx.x;
  const float* xr = x + (long)row * 1024;
  float s = 0.0f, s2 = 0.0f;
  for (int d = threadIdx.x; d < 1024; d += 256) {
    const float v = xr[d];
    s += v; s2 += v * v;
  }
  __shared__ float rs[8], rs2[8];
  #pragma unroll
  for (int m = 16; m; m >>= 1) { s += __shfl_xor(s, m); s2 += __shfl_xor(s2, m); }
  const int wv = threadIdx.x >> 5, ln = threadIdx.x & 31;
  if (ln == 0) { rs[wv] = s; rs2[wv] = s2; }
  __syncthreads();
  if (threadIdx.x == 0) {
    float a = 0.0f, c = 0.0f;
    for (int i = 0; i < 8; ++i) { a += rs[i]; c += rs2[i]; }
    rs[0] = a; rs2[0] = c;
  }
  __syncthreads();
  const float mean = rs[0] * (1.0f / 1024.0f);
  const float var = rs2[0] * (1.0f / 1024.0f) - mean * mean;
  const float inv = rsqrtf(var + 1e-5f);
  for (int d = threadIdx.x; d < 1024; d += 256)
    y[(long)row * 1024 + d] = (xr[d] - mean) * inv * g[d] + b[d];
}

__global__ void vadd(float* __restrict__ a, const float* __restrict__ b) {
  const int i = blockIdx.x * blockDim.x + threadIdx.x;
  a[i] += b[i];
}

// self-attention over the 2 query tokens (per batch x head thread)
__global__ void self_attn2(const float* __restrict__ proj, float* __restrict__ out) {
  const int id = blockIdx.x * blockDim.x + threadIdx.x;
  if (id >= 32 * 16) return;
  const int b = id >> 4, h = id & 15;
  const float* base = proj + (long)b * 2 * 3072 + h * 64;
  const float* q0 = base;            const float* q1 = base + 3072;
  const float* k0 = base + 1024;     const float* k1 = base + 3072 + 1024;
  const float* v0 = base + 2048;     const float* v1 = base + 3072 + 2048;
  float s00 = 0, s01 = 0, s10 = 0, s11 = 0;
  for (int d = 0; d < 64; ++d) {
    const float a0 = q0[d], a1 = q1[d], c0 = k0[d], c1 = k1[d];
    s00 += a0 * c0; s01 += a0 * c1; s10 += a1 * c0; s11 += a1 * c1;
  }
  const float sc = 0.125f;  // 1/sqrt(64)
  s00 *= sc; s01 *= sc; s10 *= sc; s11 *= sc;
  const float m0 = fmaxf(s00, s01), m1 = fmaxf(s10, s11);
  const float e00 = __expf(s00 - m0), e01 = __expf(s01 - m0);
  const float e10 = __expf(s10 - m1), e11 = __expf(s11 - m1);
  const float i0 = 1.0f / (e00 + e01), i1 = 1.0f / (e10 + e11);
  const float a00 = e00 * i0, a01 = e01 * i0, a10 = e10 * i1, a11 = e11 * i1;
  float* o0 = out + (long)b * 2 * 1024 + h * 64;
  float* o1 = o0 + 1024;
  for (int d = 0; d < 64; ++d) {
    o0[d] = a00 * v0[d] + a01 * v1[d];
    o1[d] = a10 * v0[d] + a11 * v1[d];
  }
}

// combine NSEG split-softmax partials -> c_attn[b][h][d]
__global__ void combine_partials(const float* __restrict__ pc, const float* __restrict__ pm,
                                 const float* __restrict__ ps, float* __restrict__ cattn) {
  const int bh = blockIdx.x;
  const int b = bh >> 4, h = bh & 15;
  float M = -1e30f;
  #pragma unroll
  for (int s = 0; s < NSEG; ++s) M = fmaxf(M, pm[(long)(b * NSEG + s) * 16 + h]);
  float wseg[NSEG];
  float tot = 0.0f;
  #pragma unroll
  for (int s = 0; s < NSEG; ++s) {
    const float w = __expf(pm[(long)(b * NSEG + s) * 16 + h] - M);
    wseg[s] = w;
    tot += ps[(long)(b * NSEG + s) * 16 + h] * w;
  }
  const float inv = 1.0f / tot;
  for (int d = threadIdx.x; d < 1024; d += blockDim.x) {
    float acc = 0.0f;
    #pragma unroll
    for (int s = 0; s < NSEG; ++s)
      acc += pc[((long)(b * NSEG + s) * 16 + h) * 1024 + d] * wseg[s];
    cattn[((long)b * 16 + h) * 1024 + d] = acc * inv;
  }
}

// ---------------------------------------------------------------------------
extern "C" void kernel_launch(void* const* d_in, const int* in_sizes, int n_in,
                              void* d_out, int out_size, void* d_ws, size_t ws_size,
                              hipStream_t stream) {
  (void)in_sizes; (void)n_in; (void)out_size; (void)ws_size;

  const float* frame   = (const float*)d_in[0];
  const float* kvs     = (const float*)d_in[1];
  const float* maxini  = (const float*)d_in[2];
  const float* qbase   = (const float*)d_in[3];
  const float* rolew   = (const float*)d_in[4];
  const float* timew   = (const float*)d_in[5];
  const float* ln1g    = (const float*)d_in[6];
  const float* ln1b    = (const float*)d_in[7];
  const float* sa_in_w = (const float*)d_in[8];
  const float* sa_in_b = (const float*)d_in[9];
  const float* sa_ow   = (const float*)d_in[10];
  const float* sa_ob   = (const float*)d_in[11];
  const float* ln2g    = (const float*)d_in[12];
  const float* ln2b    = (const float*)d_in[13];
  const float* cg_in_w = (const float*)d_in[14];
  const float* cg_in_b = (const float*)d_in[15];
  const float* cg_ow   = (const float*)d_in[16];
  const float* cg_ob   = (const float*)d_in[17];
  const float* cs_in_w = (const float*)d_in[18];
  const float* cs_in_b = (const float*)d_in[19];
  const float* cs_ow   = (const float*)d_in[20];
  const float* cs_ob   = (const float*)d_in[21];
  const float* ln3g    = (const float*)d_in[22];
  const float* ln3b    = (const float*)d_in[23];
  const float* f1w     = (const float*)d_in[24];
  const float* f1b     = (const float*)d_in[25];
  const float* f2w     = (const float*)d_in[26];
  const float* f2b     = (const float*)d_in[27];
  const float* outg    = (const float*)d_in[28];
  const float* outb    = (const float*)d_in[29];
  const int*   fidx    = (const int*)d_in[30];

  // workspace carve-up (floats)
  float* w0    = (float*)d_ws;
  float* pmean = w0;                   // 32*8*1024          = 262144
  float* q     = pmean + 262144;       // 64*1024            =  65536
  float* xbuf  = q + 65536;            // 64*1024            =  65536
  float* proj  = xbuf + 65536;         // 64*3072            = 196608
  float* sao   = proj + 196608;        // 64*1024            =  65536
  float* tmp   = sao + 65536;          // 64*1024            =  65536
  float* qp    = tmp + 65536;          // 32*1024            =  32768
  float* obuf  = qp + 32768;           // 32*1024            =  32768
  float* cattn = obuf + 32768;         // 32*16*1024         = 524288
  float* hbuf  = cattn + 524288;       // 64*4096            = 262144
  float* pm    = hbuf + 262144;        // 32*NSEG*16         =   8192
  float* ps    = pm + 8192;            // 32*NSEG*16         =   8192
  float* pc    = ps + 8192;            // 32*NSEG*16*1024    = 8388608
  unsigned short* qkbf = (unsigned short*)(pc + 8388608);  // 32*16*1024 bf16

  const long D = 1024, D2 = D * D, D3 = 3 * D2, DF = 4 * D2;

  auto gemm = [&](const float* A, int lda, long aZ,
                  const float* W, int ldw, long wZ, int transW,
                  const float* bias, int bZ,
                  float* Cf, unsigned short* Cb, int ldc, long cZ,
                  int M, int N, int K, int Z, float alpha, int gelu) {
    dim3 g(M / 16, N / 16, Z);
    gemm_wmma<<<g, 32, 0, stream>>>(A, lda, aZ, W, ldw, wZ, transW, bias, bZ,
                                    Cf, Cb, ldc, cZ, K, alpha, gelu);
  };

  // q initialization: mean token + max_init token + embeds
  mean_partial<<<dim3(32, 8, 4), 256, 0, stream>>>(frame, pmean);
  q_init<<<32, 256, 0, stream>>>(pmean, maxini, qbase, rolew, timew, fidx, q);

  for (int l = 0; l < 2; ++l) {
    const float* siw = sa_in_w + (long)l * D3;  const float* sib = sa_in_b + (long)l * 3 * D;
    const float* sow = sa_ow   + (long)l * D2;  const float* sob = sa_ob   + (long)l * D;
    const float* f1wl = f1w + (long)l * DF;     const float* f1bl = f1b + (long)l * 4 * D;
    const float* f2wl = f2w + (long)l * DF;     const float* f2bl = f2b + (long)l * D;

    // ---- self-attention over the 2 tokens
    ln_rows<<<64, 256, 0, stream>>>(q, ln1g + l * D, ln1b + l * D, xbuf);
    gemm(xbuf, 1024, 0, siw, 1024, 0, 1, sib, 0, proj, nullptr, 3072, 0,
         64, 3072, 1024, 1, 1.0f, 0);
    self_attn2<<<2, 256, 0, stream>>>(proj, sao);
    gemm(sao, 1024, 0, sow, 1024, 0, 1, sob, 0, tmp, nullptr, 1024, 0,
         64, 1024, 1024, 1, 1.0f, 0);
    vadd<<<256, 256, 0, stream>>>(q, tmp);

    // ---- cross attention: token0 -> frame_tokens (cg), token1 -> kv_salient (cs)
    ln_rows<<<64, 256, 0, stream>>>(q, ln2g + l * D, ln2b + l * D, xbuf);
    for (int tok = 0; tok < 2; ++tok) {
      const float* ciw = (tok ? cs_in_w : cg_in_w) + (long)l * D3;
      const float* cib = (tok ? cs_in_b : cg_in_b) + (long)l * 3 * D;
      const float* cow = (tok ? cs_ow : cg_ow) + (long)l * D2;
      const float* cob = (tok ? cs_ob : cg_ob) + (long)l * D;
      const float* Xp = tok ? kvs : frame;
      const int Ntok = tok ? 1024 : 4096;

      // qp = qn[:,tok] @ wq^T + bq                       (32 x 1024)
      gemm(xbuf + tok * 1024, 2048, 0, ciw, 1024, 0, 1, cib, 0,
           qp, nullptr, 1024, 0, 32, 1024, 1024, 1, 1.0f, 0);
      // QK[b,h,:] = scale * qp_h @ wk_h   (per head z; bf16 out, softmax-const dropped)
      gemm(qp, 1024, 64, ciw + D2, 1024, 64 * 1024, 0, nullptr, 0,
           nullptr, qkbf, 16384, 1024, 32, 1024, 64, 16, 0.125f, 0);
      // fused flash streaming pass over the KV source
      cross_stream<<<dim3(32, NSEG), 256, 0, stream>>>(Xp, Ntok, qkbf, pc, pm, ps);
      combine_partials<<<512, 256, 0, stream>>>(pc, pm, ps, cattn);
      // o[:, h*64:+64] = c_h @ wv_h^T + bv_h             (per head z)
      gemm(cattn, 16384, 1024, ciw + 2 * D2, 1024, 64 * 1024, 1, cib + 2048, 64,
           obuf, nullptr, 1024, 64, 32, 64, 1024, 16, 1.0f, 0);
      // out projection into the token slot of tmp
      gemm(obuf, 1024, 0, cow, 1024, 0, 1, cob, 0,
           tmp + tok * 1024, nullptr, 2048, 0, 32, 1024, 1024, 1, 1.0f, 0);
    }
    vadd<<<256, 256, 0, stream>>>(q, tmp);

    // ---- FFN
    ln_rows<<<64, 256, 0, stream>>>(q, ln3g + l * D, ln3b + l * D, xbuf);
    gemm(xbuf, 1024, 0, f1wl, 1024, 0, 1, f1bl, 0, hbuf, nullptr, 4096, 0,
         64, 4096, 1024, 1, 1.0f, 1 /*gelu*/);
    gemm(hbuf, 4096, 0, f2wl, 4096, 0, 1, f2bl, 0, tmp, nullptr, 1024, 0,
         64, 1024, 4096, 1, 1.0f, 0);
    vadd<<<256, 256, 0, stream>>>(q, tmp);
  }

  // final layernorm -> output (B,2,D) fp32
  ln_rows<<<64, 256, 0, stream>>>(q, outg, outb, (float*)d_out);
}
